// SelfAttention_70909910056986
// MI455X (gfx1250) — matrix operand order repaired and checked by reading, not compiled, not run on hardware
//
#include <hip/hip_runtime.h>
#include <hip/hip_bf16.h>

typedef __bf16 bf16;
typedef __attribute__((ext_vector_type(16))) __bf16 bf16x16;
typedef __attribute__((ext_vector_type(8)))  __bf16 bf16x8;
typedef __attribute__((ext_vector_type(4)))  __bf16 bf16x4;
typedef __attribute__((ext_vector_type(8)))  float   f32x8;
typedef __attribute__((ext_vector_type(4)))  float   f32x4;

static constexpr int Bb = 2, Ss = 2048, Ee = 768, Nh = 12, Hh = 64;
static constexpr int Kdim = 768;

// ---- fp32 -> bf16 (round to nearest even) ----
__device__ __forceinline__ bf16 to_bf16(float f) {
    union { float f; unsigned u; } c{f};
    unsigned r = c.u + 0x7fffu + ((c.u >> 16) & 1u);
    unsigned short h = (unsigned short)(r >> 16);
    return __builtin_bit_cast(__bf16, h);
}

__device__ __forceinline__ f32x8 wmma_bf16(bf16x16 a, bf16x16 b, f32x8 c) {
    return __builtin_amdgcn_wmma_f32_16x16x32_bf16(
        false, a, false, b, (short)0, c, false, false);
}

// A-fragment: 16x32 tile at p, row stride ld (elements), K contiguous.
__device__ __forceinline__ bf16x16 frag_a(const bf16* p, int ld) {
    const int lane = threadIdx.x & 31;
    const bf16* r = p + (lane & 15) * ld + ((lane >> 4) << 3);
    bf16x8 lo = *(const bf16x8*)r;
    bf16x8 hi = *(const bf16x8*)(r + 16);
    bf16x16 o;
#pragma unroll
    for (int i = 0; i < 8; ++i) { o[i] = lo[i]; o[i + 8] = hi[i]; }
    return o;
}

// B-fragment from K-transposed buffer p[n][k], row stride ld.
__device__ __forceinline__ bf16x16 frag_bt(const bf16* p, int ld) {
    const int lane = threadIdx.x & 31;
    const bf16* r = p + (lane & 15) * ld + ((lane >> 4) << 3);
    bf16x8 lo = *(const bf16x8*)r;
    bf16x8 hi = *(const bf16x8*)(r + 16);
    bf16x16 o;
#pragma unroll
    for (int i = 0; i < 8; ++i) { o[i] = lo[i]; o[i + 8] = hi[i]; }
    return o;
}

// ---------------- fp32 -> bf16 bulk convert ----------------
__global__ __launch_bounds__(256)
void cvt_f32_bf16(const float* __restrict__ in, bf16* __restrict__ out, int n4) {
    int i = blockIdx.x * blockDim.x + threadIdx.x;
    if (i >= n4) return;
    f32x4 v = *(const f32x4*)(in + 4 * (size_t)i);
    bf16x4 o;
#pragma unroll
    for (int j = 0; j < 4; ++j) o[j] = to_bf16(v[j]);
    *(bf16x4*)(out + 4 * (size_t)i) = o;
}

// ---------------- GEMM: block 256x64, wave tile 64x32, BK=32 ----------------
// MODE 0: QKV (Nout=2304) -> scatter Q(B,N,S,H), K(B,N,S,H), Vt(B,N,H,S) bf16
// MODE 1: proj (Nout=768) -> fp32 + bias to Fo
template <int MODE>
__global__ __launch_bounds__(256)
void gemm_kernel(const bf16* __restrict__ A, const bf16* __restrict__ W,
                 const float* __restrict__ bias,
                 bf16* __restrict__ Qo, bf16* __restrict__ Ko, bf16* __restrict__ Vo,
                 float* __restrict__ Fo, int Nout) {
    constexpr int BM = 256, BN = 64, BK = 32, LDT = 40;   // 80B rows: 16B aligned
    __shared__ __align__(16) bf16 As[BM * LDT];
    __shared__ __align__(16) bf16 Bs[BN * LDT];
    const int tid  = threadIdx.x;
    const int wave = tid >> 5;
    const int wr   = wave >> 1, wc = wave & 1;            // 4x2 waves, 64x32 each
    const int m0   = blockIdx.y * BM;
    const int n0   = blockIdx.x * BN;

    f32x8 acc[4][2] = {};

    for (int kb = 0; kb < Kdim; kb += BK) {
        {   // stage A tile 256x32: one full row (32 elems) per thread
            const bf16* g = A + (size_t)(m0 + tid) * Kdim + kb;
#pragma unroll
            for (int u = 0; u < 4; ++u)
                *(bf16x8*)&As[tid * LDT + u * 8] = *(const bf16x8*)(g + u * 8);
        }
        {   // stage W tile transposed -> Bs[n][k], 64x32: 8 elems / thread
            int nl = tid >> 2;
            int kc = (tid & 3) * 8;
#pragma unroll
            for (int k = 0; k < 8; ++k)
                Bs[nl * LDT + kc + k] = W[(size_t)(kb + kc + k) * Nout + n0 + nl];
        }
        __syncthreads();

        bf16x16 bfr0 = frag_bt(&Bs[(wc * 32) * LDT], LDT);
        bf16x16 bfr1 = frag_bt(&Bs[(wc * 32 + 16) * LDT], LDT);
#pragma unroll
        for (int i = 0; i < 4; ++i) {
            bf16x16 afr = frag_a(&As[(wr * 64 + i * 16) * LDT], LDT);
            acc[i][0] = wmma_bf16(afr, bfr0, acc[i][0]);
            acc[i][1] = wmma_bf16(afr, bfr1, acc[i][1]);
        }
        __syncthreads();
    }

    const int lane = tid & 31;
    const int nl = lane & 15, hf = lane >> 4;
#pragma unroll
    for (int i = 0; i < 4; ++i)
#pragma unroll
        for (int j = 0; j < 2; ++j) {
            int row0 = m0 + wr * 64 + i * 16 + hf * 8;
            int col  = n0 + wc * 32 + j * 16 + nl;
            float bv = bias[col];
#pragma unroll
            for (int r = 0; r < 8; ++r) {
                int m = row0 + r;
                float v = acc[i][j][r] + bv;
                if (MODE == 0) {
                    int b = m >> 11, s = m & (Ss - 1);
                    int sel = (col >= 1536) ? 2 : (col >= 768 ? 1 : 0);
                    int e = col - sel * 768;
                    int nh = e >> 6, h = e & (Hh - 1);
                    int bn = b * Nh + nh;
                    if (sel == 0)      Qo[((size_t)bn * Ss + s) * Hh + h] = to_bf16(v);
                    else if (sel == 1) Ko[((size_t)bn * Ss + s) * Hh + h] = to_bf16(v);
                    else               Vo[((size_t)bn * Hh + h) * Ss + s] = to_bf16(v);
                } else {
                    Fo[(size_t)m * Nout + col] = v;
                }
            }
        }
}

// ------- async stage of one 64x64 K tile + 64x64 Vt tile into LDS (stride 72) -------
__device__ __forceinline__ void async_copy_tile(const bf16* __restrict__ gK,
                                                const bf16* __restrict__ gV,
                                                bf16* lK, bf16* lV, int tid) {
#pragma unroll
    for (int i = 0; i < 2; ++i) {
        int c = tid + 256 * i;               // 512 chunks of 16B per tile
        int row = c >> 3, col = (c & 7) * 8;
        unsigned lk = (unsigned)(size_t)(lK + row * 72 + col);
        unsigned long long gk = (unsigned long long)(size_t)(gK + (size_t)row * Hh + col);
        asm volatile("global_load_async_to_lds_b128 %0, %1, off"
                     :: "v"(lk), "v"(gk) : "memory");
        unsigned lv = (unsigned)(size_t)(lV + row * 72 + col);
        unsigned long long gv = (unsigned long long)(size_t)(gV + (size_t)row * Ss + col);
        asm volatile("global_load_async_to_lds_b128 %0, %1, off"
                     :: "v"(lv), "v"(gv) : "memory");
    }
}

// ---------------- Flash attention: 128 Q rows/block, KV tiles of 64, double-buffered ----------------
__global__ __launch_bounds__(256)
void attn_kernel(const bf16* __restrict__ Q, const bf16* __restrict__ K,
                 const bf16* __restrict__ Vt, bf16* __restrict__ O) {
    constexpr int QBLK = 128, LD = 72;                 // 144B rows: 16B aligned
    constexpr int TILE_BYTES = 64 * LD * 2;            // 9216 per matrix
    constexpr int BUF_BYTES  = 2 * TILE_BYTES;         // K tile + V tile per buffer
    extern __shared__ __align__(16) char smem[];       // 2 buffers + 8*P = 55296 B

    const int tid  = threadIdx.x;
    const int wave = tid >> 5;
    const int lane = tid & 31;
    const int nl = lane & 15, hf = lane >> 4;

    const int nblk = Ss / QBLK;                        // 16
    const int bn = blockIdx.x / nblk;
    const int qb = blockIdx.x % nblk;
    const int b = bn / Nh, nh = bn % Nh;

    const bf16* Qh = Q  + (size_t)bn * Ss * Hh;
    const bf16* Kh = K  + (size_t)bn * Ss * Hh;
    const bf16* Vh = Vt + (size_t)bn * Hh * Ss;
    const int q0 = qb * QBLK + wave * 16;

    bf16* Pw = (bf16*)(smem + 2 * BUF_BYTES) + (size_t)wave * 16 * LD;

    bf16x16 aq0 = frag_a(Qh + (size_t)q0 * Hh, Hh);
    bf16x16 aq1 = frag_a(Qh + (size_t)q0 * Hh + 32, Hh);

    f32x8 o[4] = {};
    float mrow[8], lrow[8];
#pragma unroll
    for (int r = 0; r < 8; ++r) { mrow[r] = -1e30f; lrow[r] = 0.f; }

    const int nkv = 2 * (qb + 1);                      // causal extent in 64-wide tiles
    // prefetch tile 0 into buffer 0
    async_copy_tile(Kh, Vh, (bf16*)smem, (bf16*)(smem + TILE_BYTES), tid);

    for (int kt = 0; kt < nkv; ++kt) {
        const int cur = kt & 1;
        const int sb = kt * 64;
        asm volatile("s_wait_asynccnt 0x0" ::: "memory");
        __syncthreads();                               // tile ready; all done with 'other' buffer
        if (kt + 1 < nkv) {                            // stream next tile during compute
            char* nb = smem + (1 - cur) * BUF_BYTES;
            async_copy_tile(Kh + (size_t)(kt + 1) * 64 * Hh, Vh + (kt + 1) * 64,
                            (bf16*)nb, (bf16*)(nb + TILE_BYTES), tid);
        }

        if (sb <= q0 + 15) {                           // wave-uniform causal skip
            const bf16* Ksb = (const bf16*)(smem + cur * BUF_BYTES);
            const bf16* Vsb = (const bf16*)(smem + cur * BUF_BYTES + TILE_BYTES);
            f32x8 s[4];
#pragma unroll
            for (int j = 0; j < 4; ++j) {
                f32x8 t{};
                t = wmma_bf16(aq0, frag_bt(&Ksb[(j * 16) * LD], LD), t);
                t = wmma_bf16(aq1, frag_bt(&Ksb[(j * 16) * LD + 32], LD), t);
                s[j] = t;
            }
#pragma unroll
            for (int r = 0; r < 8; ++r) {              // scale + causal mask
                int row = q0 + r + 8 * hf;
#pragma unroll
                for (int j = 0; j < 4; ++j) {
                    int c = sb + j * 16 + nl;
                    float x = s[j][r] * 0.125f;        // 1/sqrt(64)
                    s[j][r] = (c > row) ? -1e30f : x;
                }
            }
#pragma unroll
            for (int r = 0; r < 8; ++r) {              // online softmax per row
                float v = fmaxf(fmaxf(s[0][r], s[1][r]), fmaxf(s[2][r], s[3][r]));
                v = fmaxf(v, __shfl_xor(v, 1, 32));
                v = fmaxf(v, __shfl_xor(v, 2, 32));
                v = fmaxf(v, __shfl_xor(v, 4, 32));
                v = fmaxf(v, __shfl_xor(v, 8, 32));
                float mnew  = fmaxf(mrow[r], v);
                float alpha = __expf(mrow[r] - mnew);
                float p0 = __expf(s[0][r] - mnew), p1 = __expf(s[1][r] - mnew);
                float p2 = __expf(s[2][r] - mnew), p3 = __expf(s[3][r] - mnew);
                float ps = (p0 + p1) + (p2 + p3);
                ps += __shfl_xor(ps, 1, 32);
                ps += __shfl_xor(ps, 2, 32);
                ps += __shfl_xor(ps, 4, 32);
                ps += __shfl_xor(ps, 8, 32);
                lrow[r] = lrow[r] * alpha + ps;
                mrow[r] = mnew;
                o[0][r] *= alpha; o[1][r] *= alpha; o[2][r] *= alpha; o[3][r] *= alpha;
                int m = r + 8 * hf;                    // D-layout -> P[m][s_local]
                Pw[m * LD +      nl] = to_bf16(p0);
                Pw[m * LD + 16 + nl] = to_bf16(p1);
                Pw[m * LD + 32 + nl] = to_bf16(p2);
                Pw[m * LD + 48 + nl] = to_bf16(p3);
            }
            // O += P(16x64) @ V(64x64)
            bf16x16 ap0 = frag_a(Pw, LD);
            bf16x16 ap1 = frag_a(Pw + 32, LD);
#pragma unroll
            for (int j = 0; j < 4; ++j) {
                o[j] = wmma_bf16(ap0, frag_bt(&Vsb[(j * 16) * LD], LD), o[j]);
                o[j] = wmma_bf16(ap1, frag_bt(&Vsb[(j * 16) * LD + 32], LD), o[j]);
            }
        }
        __syncthreads();                               // done reading buf[cur] before reuse
    }

#pragma unroll
    for (int r = 0; r < 8; ++r) {                      // write O into (B,S,E) bf16
        int row = q0 + r + 8 * hf;
        float inv = 1.0f / lrow[r];
        bf16* orow = O + ((size_t)b * Ss + row) * Ee + nh * Hh;
        orow[0 * 16 + nl] = to_bf16(o[0][r] * inv);
        orow[1 * 16 + nl] = to_bf16(o[1][r] * inv);
        orow[2 * 16 + nl] = to_bf16(o[2][r] * inv);
        orow[3 * 16 + nl] = to_bf16(o[3][r] * inv);
    }
}

// ------------------------------- launcher -------------------------------
extern "C" void kernel_launch(void* const* d_in, const int* in_sizes, int n_in,
                              void* d_out, int out_size, void* d_ws, size_t ws_size,
                              hipStream_t stream) {
    // inputs: 0=mask(unused, causal handled analytically), 1=x, 2=Wqkv, 3=bqkv, 4=Wproj, 5=bproj
    const float* x     = (const float*)d_in[1];
    const float* Wqkv  = (const float*)d_in[2];
    const float* bqkv  = (const float*)d_in[3];
    const float* Wproj = (const float*)d_in[4];
    const float* bproj = (const float*)d_in[5];
    float* out = (float*)d_out;

    const size_t nX  = (size_t)Bb * Ss * Ee;
    const size_t nWq = (size_t)Ee * 3 * Ee;
    const size_t nWp = (size_t)Ee * Ee;
    const size_t nH  = (size_t)Bb * Nh * Ss * Hh;

    char* p = (char*)d_ws;
    auto take = [&](size_t elems) { bf16* q = (bf16*)p; p += ((elems * 2 + 255) & ~255ull); return q; };
    bf16* xbf  = take(nX);
    bf16* Wqbf = take(nWq);
    bf16* Wpbf = take(nWp);
    bf16* Qbf  = take(nH);
    bf16* Kbf  = take(nH);
    bf16* Vtbf = take(nH);
    bf16* Obf  = take(nX);

    cvt_f32_bf16<<<(int)(nX  / 4 / 256), 256, 0, stream>>>(x,     xbf,  (int)(nX  / 4));
    cvt_f32_bf16<<<(int)(nWq / 4 / 256), 256, 0, stream>>>(Wqkv,  Wqbf, (int)(nWq / 4));
    cvt_f32_bf16<<<(int)(nWp / 4 / 256), 256, 0, stream>>>(Wproj, Wpbf, (int)(nWp / 4));

    // QKV GEMM: M=4096, N=2304, K=768
    gemm_kernel<0><<<dim3(2304 / 64, 4096 / 256), 256, 0, stream>>>(
        xbf, Wqbf, bqkv, Qbf, Kbf, Vtbf, nullptr, 3 * Ee);

    // attention: B*N*(S/128) = 384 blocks, 55296B dynamic LDS
    attn_kernel<<<Bb * Nh * (Ss / 128), 256, 4 * (64 * 72 * 2) + 8 * (16 * 72 * 2), stream>>>(
        Qbf, Kbf, Vtbf, Obf);

    // output projection: M=4096, N=768, K=768
    gemm_kernel<1><<<dim3(768 / 64, 4096 / 256), 256, 0, stream>>>(
        Obf, Wpbf, bproj, nullptr, nullptr, nullptr, out, Ee);
}
